// LSTMModel_71408126263352
// MI455X (gfx1250) — matrix operand order repaired and checked by reading, not compiled, not run on hardware
//
#include <hip/hip_runtime.h>
#include <hip/hip_bf16.h>
#include <math.h>

typedef __attribute__((ext_vector_type(16))) _Float16 v16h;
typedef __attribute__((ext_vector_type(8)))  _Float16 v8h;
typedef __attribute__((ext_vector_type(8)))  float    v8f;

#define HDIM   64     // hidden per direction
#define GDIM   256    // 4*H gate width
#define BT     16     // batch rows per workgroup (one M tile)
#define APITCH 192    // halves per row of LDS A-buffer (x cols | h cols)
#define HOFF   128    // h columns start in A-buffer

#define LOG2E  1.44269504088896340736f

// Branch-free fast activations using hardware TRANS32 ops (co-execute with VALU,
// no exec-mask divergence on the sequential critical path).
__device__ __forceinline__ float fast_sigmoid(float x) {
    // 1 / (1 + 2^(-x*log2(e))) : v_exp_f32 + v_rcp_f32
    return __builtin_amdgcn_rcpf(1.0f + __builtin_amdgcn_exp2f(-LOG2E * x));
}

#if defined(__has_builtin) && __has_builtin(__builtin_amdgcn_tanhf)
__device__ __forceinline__ float fast_tanh(float x) { return __builtin_amdgcn_tanhf(x); }
#else
__device__ __forceinline__ float fast_tanh(float x) {
    // tanh(x) = 2*sigmoid(2x) - 1, branch-free
    float e = __builtin_amdgcn_exp2f(-2.0f * LOG2E * x);
    return __builtin_amdgcn_rcpf(1.0f + e) * 2.0f - 1.0f;
}
#endif

// A fragment (16x32 f16, MxK) from LDS row-major [16][APITCH]:
// lanes 0-15: row=lane, halves = K{kb..kb+7, kb+16..kb+23}
// lanes 16-31: row=lane-16, halves = K{kb+8..kb+15, kb+24..kb+31}
__device__ __forceinline__ v16h load_a_frag(const _Float16* a_buf, int lane, int kbase) {
    int row = lane & 15;
    int k0  = kbase + ((lane >> 4) & 1) * 8;
    const _Float16* p = a_buf + row * APITCH + k0;
    v8h lo = *(const v8h*)(p);
    v8h hi = *(const v8h*)(p + 16);
    v16h a;
#pragma unroll
    for (int i = 0; i < 8; i++) { a[i] = lo[i]; a[8 + i] = hi[i]; }
    return a;
}

// B fragment (32x16 f16, KxN). B(k,n) = W[n][k] (W row-major [GDIM][ld], f32->f16).
// lanes 0-15 hold K={kb..kb+7, kb+16..kb+23}, lanes 16-31 hold K={kb+8..kb+15, kb+24..kb+31}; N = nbase + (lane&15).
__device__ __forceinline__ v16h load_b_frag(const float* w, int ld, int lane, int nbase, int kbase) {
    int n  = nbase + (lane & 15);
    int k0 = kbase + ((lane >> 4) & 1) * 8;
    const float* p = w + (size_t)n * ld + k0;
    v16h b;
#pragma unroll
    for (int i = 0; i < 16; i++) b[i] = (_Float16)p[i < 8 ? i : i + 8];
    return b;
}

// Persistent per-(direction, batch-tile) LSTM scan.
// DPATH==0: layer 0 (input size 1): x-term folded as a rank-1 K=32 WMMA chunk
//           (x[row] in A col 0, w_ih[n] in B row 0, rest zeros).
// DPATH==1: layers 1..3 (input size 128, f16 activations, input GEMM = 4 WMMA chunks)
template <int DPATH>
__global__ __launch_bounds__(256) void lstm_scan(
    const float* __restrict__ x0,        // DPATH==0: [B][T][1] fp32
    const _Float16* __restrict__ xin,    // DPATH==1: [B][T][128] f16
    const float* __restrict__ w_ih_base, // [2][GDIM][D]
    const float* __restrict__ w_hh_base, // [2][GDIM][HDIM]
    const float* __restrict__ b_base,    // [2][GDIM]
    _Float16* __restrict__ hout,         // [B][T][128] f16 (fwd cols 0-63, bwd 64-127)
    int T) {
    const int NX = DPATH ? 4 : 1; // number of K=32 input chunks
    const int tid  = threadIdx.x;
    const int lane = tid & 31;
    const int wave = tid >> 5;
    const int dir   = blockIdx.x >> 4;        // 0 = fwd, 1 = bwd
    const int bbase = (blockIdx.x & 15) * BT; // batch tile

    const int Din = DPATH ? 128 : 1;
    const float* w_ih = w_ih_base + (size_t)dir * GDIM * Din;
    const float* w_hh = w_hh_base + (size_t)dir * GDIM * HDIM;
    const float* bias = b_base + (size_t)dir * GDIM;

    __shared__ _Float16 a_buf[BT * APITCH]; // f16 A operand staging (x | h)
    __shared__ float    g_buf[BT * GDIM];   // f32 gate pre-activations

    // ---- loop-invariant B fragments live in VGPRs for the whole scan ----
    const int n0 = wave * 32; // this wave's two N tiles: n0, n0+16
    v16h bh[2][2];
#pragma unroll
    for (int tSel = 0; tSel < 2; tSel++)
#pragma unroll
        for (int kc = 0; kc < 2; kc++)
            bh[tSel][kc] = load_b_frag(w_hh, HDIM, lane, n0 + tSel * 16, kc * 32);

    v16h bx[2][4];
#pragma unroll
    for (int tSel = 0; tSel < 2; tSel++) {
        if (DPATH) {
#pragma unroll
            for (int kc = 0; kc < 4; kc++)
                bx[tSel][kc] = load_b_frag(w_ih, 128, lane, n0 + tSel * 16, kc * 32);
        } else {
            // rank-1 B fragment: only K=0 row nonzero (lanes 0-15, half 0)
            const int n = n0 + tSel * 16 + (lane & 15);
            v16h bv;
#pragma unroll
            for (int i = 0; i < 16; i++) bv[i] = (_Float16)0.f;
            bv[0] = (lane & 16) ? (_Float16)0.f : (_Float16)w_ih[n];
            bx[tSel][0] = bv;
        }
    }

    // loop-invariant C-init scalars (f32 C/D layout: N = nbase+(lane&15), M = rb+r)
    const int colA = n0 + (lane & 15);
    const int colB = colA + 16;
    const int rb   = (lane & 16) ? 8 : 0;
    const float biasA = bias[colA], biasB = bias[colB];

    // cell state in registers: thread owns hidden unit j for 4 batch rows
    const int j  = tid & 63;
    const int r0 = (tid >> 6) * 4;
    float cst[4] = {0.f, 0.f, 0.f, 0.f};

    // h(t=-1) = 0
#pragma unroll
    for (int r = 0; r < 4; r++) a_buf[(r0 + r) * APITCH + HOFF + j] = (_Float16)0.f;

    // DPATH==0: zero cols 1..31 of the rank-1 x chunk once (col 0 written per step)
    if (!DPATH) {
        for (int idx = tid; idx < BT * 32; idx += 256) {
            const int row = idx >> 5, col = idx & 31;
            if (col) a_buf[row * APITCH + col] = (_Float16)0.f;
        }
    }

    // ---- double-buffered input prefetch: load step-0 input up front ----
    const int srow = tid >> 4;            // DPATH==1 staging row
    const int scol = (tid & 15) * 8;      // DPATH==1 staging col (8 halves)
    v8h   xs;                             // DPATH==1: staged x tile chunk
    float xsc = 0.f;                      // DPATH==0: staged x scalar (tid < BT)
    {
        const int tt0 = dir ? (T - 1) : 0;
        if (DPATH) {
            xs = *(const v8h*)(xin + (((size_t)(bbase + srow)) * T + tt0) * 128 + scol);
        } else if (tid < BT) {
            xsc = x0[(size_t)(bbase + tid) * T + tt0];
        }
    }

    for (int t = 0; t < T; t++) {
        const int tt = dir ? (T - 1 - t) : t;

        // commit prefetched x_tt tile
        if (DPATH) {
            *(v8h*)(a_buf + srow * APITCH + scol) = xs;
        } else if (tid < BT) {
            a_buf[tid * APITCH] = (_Float16)xsc;
        }
        __syncthreads(); // x staged, h(t-1) visible

        // issue next step's global input load NOW; latency hides behind WMMA+cell
        const int tn  = (t + 1 < T) ? (t + 1) : t;
        const int ttn = dir ? (T - 1 - tn) : tn;
        v8h   xs_n;
        float xsc_n = 0.f;
        if (DPATH) {
            xs_n = *(const v8h*)(xin + (((size_t)(bbase + srow)) * T + ttn) * 128 + scol);
        } else if (tid < BT) {
            xsc_n = x0[(size_t)(bbase + tid) * T + ttn];
        }

        // A fragments (shared between both N tiles of this wave)
        v16h ah0 = load_a_frag(a_buf, lane, HOFF);
        v16h ah1 = load_a_frag(a_buf, lane, HOFF + 32);
        v16h ax[4];
#pragma unroll
        for (int kc = 0; kc < NX; kc++) ax[kc] = load_a_frag(a_buf, lane, kc * 32);

#pragma unroll
        for (int tSel = 0; tSel < 2; tSel++) {
            const float bs = tSel ? biasB : biasA;
            v8f acc;
#pragma unroll
            for (int r = 0; r < 8; r++) acc[r] = bs;
#pragma unroll
            for (int kc = 0; kc < NX; kc++)
                acc = __builtin_amdgcn_wmma_f32_16x16x32_f16(
                    false, ax[kc], false, bx[tSel][kc], (short)0, acc, false, false);
            acc = __builtin_amdgcn_wmma_f32_16x16x32_f16(
                false, ah0, false, bh[tSel][0], (short)0, acc, false, false);
            acc = __builtin_amdgcn_wmma_f32_16x16x32_f16(
                false, ah1, false, bh[tSel][1], (short)0, acc, false, false);

            const int col = tSel ? colB : colA;
#pragma unroll
            for (int r = 0; r < 8; r++) g_buf[(rb + r) * GDIM + col] = acc[r];
        }
        __syncthreads(); // gates visible

        // elementwise LSTM cell update (fp32, branch-free TRANS ops), h back as f16
#pragma unroll
        for (int r = 0; r < 4; r++) {
            const int row = r0 + r;
            float gi = g_buf[row * GDIM + j];
            float gf = g_buf[row * GDIM + 64 + j];
            float gg = g_buf[row * GDIM + 128 + j];
            float go = g_buf[row * GDIM + 192 + j];
            float c  = fast_sigmoid(gf) * cst[r] + fast_sigmoid(gi) * fast_tanh(gg);
            cst[r]   = c;
            float h  = fast_sigmoid(go) * fast_tanh(c);
            _Float16 hh = (_Float16)h;
            a_buf[row * APITCH + HOFF + j] = hh;
            hout[(((size_t)(bbase + row)) * T + tt) * 128 + dir * 64 + j] = hh;
        }

        // rotate prefetch buffers
        if (DPATH) xs = xs_n;
        else       xsc = xsc_n;
        // next iteration's first barrier orders h-writes vs WMMA reads
    }
}

// Linear(128->1) + max over T, one workgroup per batch row.
__global__ __launch_bounds__(256) void score_max(const _Float16* __restrict__ h4,
                                                 const float* __restrict__ lin_w,
                                                 const float* __restrict__ lin_b,
                                                 float* __restrict__ out, int T) {
    __shared__ float wsm[128];
    __shared__ float red[256];
    const int tid = threadIdx.x;
    const int b   = blockIdx.x;
    if (tid < 128) wsm[tid] = lin_w[tid];
    __syncthreads();
    const float lb = lin_b[0];
    float m = -3.402823466e38f;
    for (int t = tid; t < T; t += 256) {
        const _Float16* p = h4 + ((size_t)b * T + t) * 128;
        float s = 0.f;
#pragma unroll
        for (int k = 0; k < 128; k += 8) {
            v8h hv = *(const v8h*)(p + k);
#pragma unroll
            for (int i = 0; i < 8; i++) s = fmaf((float)hv[i], wsm[k + i], s);
        }
        m = fmaxf(m, s + lb);
    }
    red[tid] = m;
    __syncthreads();
    for (int s = 128; s > 0; s >>= 1) {
        if (tid < s) red[tid] = fmaxf(red[tid], red[tid + s]);
        __syncthreads();
    }
    if (tid == 0) out[b] = red[0];
}

extern "C" void kernel_launch(void* const* d_in, const int* in_sizes, int n_in,
                              void* d_out, int out_size, void* d_ws, size_t ws_size,
                              hipStream_t stream) {
    const float* x    = (const float*)d_in[0]; // [256,2048,1]
    const float* wih0 = (const float*)d_in[1]; // [2,256,1]
    const float* whh0 = (const float*)d_in[2]; // [2,256,64]
    const float* b0   = (const float*)d_in[3]; // [2,256]
    const float* wih  = (const float*)d_in[4]; // [3,2,256,128]
    const float* whh  = (const float*)d_in[5]; // [3,2,256,64]
    const float* bb   = (const float*)d_in[6]; // [3,2,256]
    const float* lw   = (const float*)d_in[7]; // [1,128]
    const float* lb   = (const float*)d_in[8]; // [1]
    float* out = (float*)d_out;                // [256]

    const int B = 256, T = 2048;
    const size_t hbytes = (size_t)B * T * 128 * sizeof(_Float16); // 128 MiB (fits L2)
    _Float16* hA = (_Float16*)d_ws;
    _Float16* hB = (_Float16*)((char*)d_ws + hbytes);

    dim3 blk(256), grd(32); // 16 batch tiles x 2 directions

    // layer 0 (input size 1)
    lstm_scan<0><<<grd, blk, 0, stream>>>(x, nullptr, wih0, whh0, b0, hA, T);

    // layers 1..3 (input size 128), ping-pong hA/hB
    _Float16* cur = hA;
    _Float16* nxt = hB;
    for (int l = 0; l < 3; l++) {
        lstm_scan<1><<<grd, blk, 0, stream>>>(nullptr, cur,
                                              wih + (size_t)l * 2 * 256 * 128,
                                              whh + (size_t)l * 2 * 256 * 64,
                                              bb + (size_t)l * 2 * 256,
                                              nxt, T);
        _Float16* tmp = cur; cur = nxt; nxt = tmp;
    }

    // final: Linear + max over time (cur == hB after 3 swaps)
    score_max<<<dim3(256), blk, 0, stream>>>(cur, lw, lb, out, T);
}
